// CausalSelfAttention_10840497455506
// MI455X (gfx1250) — compile-verified, matrix operand order checked by reading, not run on hardware
//
#include <hip/hip_runtime.h>
#include <hip/hip_bf16.h>
#include <math.h>

// ---------------------------------------------------------------------------
// CDNA5 (gfx1250) causal self-attention, bf16 WMMA pipeline.
//   x[B,T,C] f32, Wq/Wk/Wv/Wo[C,C] f32  ->  out[B,T,C] f32
//   B=4, T=2048, C=1024, H=16, D=64
// ---------------------------------------------------------------------------

typedef __attribute__((ext_vector_type(16))) __bf16        v16bf;
typedef __attribute__((ext_vector_type(8)))  float         v8f;
typedef __attribute__((ext_vector_type(4)))  unsigned int  u32x4;

#define DEV __device__ __forceinline__

constexpr int kB    = 4;
constexpr int kT    = 2048;
constexpr int kC    = 1024;
constexpr int kH    = 16;
constexpr int kD    = 64;
constexpr int kM    = kB * kT;   // 8192 rows

// round-to-nearest-even f32 -> bf16 (bit pattern)
DEV unsigned short f32_to_bf16_bits(float f) {
  union { float f; unsigned int u; } v;
  v.f = f;
  unsigned int lsb = (v.u >> 16) & 1u;
  v.u += 0x7fffu + lsb;
  return (unsigned short)(v.u >> 16);
}

// Load a 16x32 bf16 fragment in the ISA 16-bit A-matrix lane layout:
//   lanes 0-15 : row M = lane,    K = {k0+0..7,  k0+16..23}
//   lanes 16-31: row M = lane-16, K = {k0+8..15, k0+24..31}
// The WMMA B operand is loaded with the same helper from the *transposed*
// source (i.e. from W stored [out,in] row-major), which matches x @ W^T.
DEV v16bf load_frag(const unsigned short* __restrict__ base, int row0, int ld, int k0) {
  const int lane = threadIdx.x & 31;
  const int m  = lane & 15;
  const int hi = lane >> 4;
  const unsigned short* p = base + (size_t)(row0 + m) * (size_t)ld + k0 + hi * 8;
  union { v16bf v; u32x4 q[2]; } f;
  f.q[0] = *(const u32x4*)(p);
  f.q[1] = *(const u32x4*)(p + 16);
  return f.v;
}

DEV v8f wmma_bf16(v16bf a, v16bf b, v8f c) {
  // D(f32 16x16) = A(bf16 16x32) x B(bf16 32x16) + C
  return __builtin_amdgcn_wmma_f32_16x16x32_bf16(
      /*neg_a=*/false, a, /*neg_b=*/false, b,
      /*c_mod=*/(short)0, c, /*reuse_a=*/false, /*reuse_b=*/false);
}

// Lane-XOR exchange via ds_swizzle_b32 group-of-32 mode (no index VGPR):
// offset = {1'b0, xor_mask[4:0], or_mask=0, and_mask=0x1f}
template <int MASK>
DEV float xor_swizzle(float x) {
  union { float f; int i; } v;
  v.f = x;
  v.i = __builtin_amdgcn_ds_swizzle(v.i, (MASK << 10) | 0x1f);
  return v.f;
}

// Batched 16-lane-half reductions over 8 independent row chains: all 8 DS ops
// of a step are issued before any consumer, so they clause behind a single
// s_wait_dscnt instead of 32 serialized waits.
DEV void rowmax_reduce(float* cur) {
  float t[8];
#define RMAX_STEP(MASK)                                                   \
  _Pragma("unroll") for (int r = 0; r < 8; ++r) t[r] = xor_swizzle<MASK>(cur[r]); \
  _Pragma("unroll") for (int r = 0; r < 8; ++r) cur[r] = fmaxf(cur[r], t[r]);
  RMAX_STEP(1) RMAX_STEP(2) RMAX_STEP(4) RMAX_STEP(8)
#undef RMAX_STEP
}

DEV void rowsum_reduce(float* ps) {
  float t[8];
#define RSUM_STEP(MASK)                                                   \
  _Pragma("unroll") for (int r = 0; r < 8; ++r) t[r] = xor_swizzle<MASK>(ps[r]); \
  _Pragma("unroll") for (int r = 0; r < 8; ++r) ps[r] += t[r];
  RSUM_STEP(1) RSUM_STEP(2) RSUM_STEP(4) RSUM_STEP(8)
#undef RSUM_STEP
}

// ---------------------------------------------------------------------------
// f32 -> bf16 elementwise conversion (grid-stride)
// ---------------------------------------------------------------------------
__global__ void cvt_bf16_kernel(const float* __restrict__ in,
                                unsigned short* __restrict__ out, int n) {
  int i = blockIdx.x * blockDim.x + threadIdx.x;
  const int stride = gridDim.x * blockDim.x;
  for (; i < n; i += stride) out[i] = f32_to_bf16_bits(in[i]);
}

// ---------------------------------------------------------------------------
// GEMM: out = scale * (A[M,C](bf16) @ W[C,C]^T)  (W stored [out,in] row-major)
// Wave tile: 32M x 64N (8 WMMAs per 6 fragment loads), 4 waves stacked in M.
// MODE 0: f32 row-major [M,C]            (final Wo projection -> d_out)
// MODE 1: bf16 head-split [B*H, T, D]    (Q gets scale=1/sqrt(D) folded in, K)
// MODE 2: bf16 transposed  [B*H, D, T]   (V, so PV B-operand is contiguous)
// ---------------------------------------------------------------------------
template <int MODE>
__global__ __launch_bounds__(128) void gemm_kernel(
    const unsigned short* __restrict__ A,
    const unsigned short* __restrict__ W,
    void* __restrict__ out, float outScale) {
  const int wave = threadIdx.x >> 5;
  const int m0 = (blockIdx.x * 4 + wave) * 32;
  const int n0 = blockIdx.y * 64;

  v8f acc[8] = {v8f{}, v8f{}, v8f{}, v8f{}, v8f{}, v8f{}, v8f{}, v8f{}};

  for (int k0 = 0; k0 < kC; k0 += 32) {
    const v16bf a0 = load_frag(A, m0,      kC, k0);
    const v16bf a1 = load_frag(A, m0 + 16, kC, k0);
#pragma unroll
    for (int j = 0; j < 4; ++j) {
      const v16bf b = load_frag(W, n0 + j * 16, kC, k0);
      acc[j]     = wmma_bf16(a0, b, acc[j]);
      acc[4 + j] = wmma_bf16(a1, b, acc[4 + j]);
    }
  }

  const int lane = threadIdx.x & 31;
  const int nn = lane & 15;
  const int hi = lane >> 4;
#pragma unroll
  for (int part = 0; part < 2; ++part) {
#pragma unroll
    for (int j = 0; j < 4; ++j) {
#pragma unroll
      for (int r = 0; r < 8; ++r) {
        const int m = m0 + part * 16 + r + hi * 8;   // global row (b*T + t)
        const int n = n0 + j * 16 + nn;              // global col (h*D + d)
        const float val = acc[part * 4 + j][r] * outScale;
        if (MODE == 0) {
          ((float*)out)[(size_t)m * kC + n] = val;
        } else {
          const int b = m >> 11;                     // m / T
          const int t = m & (kT - 1);
          const int h = n >> 6;                      // n / D
          const int d = n & (kD - 1);
          const unsigned short bv = f32_to_bf16_bits(val);
          if (MODE == 1)
            ((unsigned short*)out)[(((size_t)(b * kH + h)) * kT + t) * kD + d] = bv;
          else
            ((unsigned short*)out)[(((size_t)(b * kH + h)) * kD + d) * kT + t] = bv;
        }
      }
    }
  }
}

// ---------------------------------------------------------------------------
// Online-softmax update for one 16x32 score tile held in D-fragment layout.
// (Q is pre-scaled by 1/sqrt(D), so no score scaling here.)
// DIAG is a template parameter: mask VALU exists only in the peeled diagonal
// chunk, never in the steady-state loop.
// ---------------------------------------------------------------------------
template <bool DIAG>
DEV void softmax_update(v8f& s0, v8f& s1, float* mrow, float* lrow, v8f* o,
                        int rowbase, int kc,
                        unsigned short* plds, int nn, int hi) {
  float cur[8];
#pragma unroll
  for (int r = 0; r < 8; ++r) {
    float a0 = s0[r];
    float a1 = s1[r];
    if (DIAG) {                                  // only the diagonal chunk
      const int m = rowbase + r + hi * 8;
      a0 = (kc + nn      <= m) ? a0 : -INFINITY;
      a1 = (kc + 16 + nn <= m) ? a1 : -INFINITY;
    }
    s0[r] = a0; s1[r] = a1;
    cur[r] = fmaxf(a0, a1);
  }
  // row-max across the 16-lane half (rows 0-7 in lanes 0-15, 8-15 in 16-31)
  rowmax_reduce(cur);

  float corr[8], ps[8];
#pragma unroll
  for (int r = 0; r < 8; ++r) {
    const float mn = fmaxf(mrow[r], cur[r]);
    corr[r] = __expf(mrow[r] - mn);
    mrow[r] = mn;
    const float p0 = __expf(s0[r] - mn);
    const float p1 = __expf(s1[r] - mn);
    ps[r] = p0 + p1;
    const int row = r + hi * 8;
    plds[row * 32 + nn]      = f32_to_bf16_bits(p0);
    plds[row * 32 + 16 + nn] = f32_to_bf16_bits(p1);
  }
  rowsum_reduce(ps);
#pragma unroll
  for (int r = 0; r < 8; ++r) {
    lrow[r] = lrow[r] * corr[r] + ps[r];
#pragma unroll
    for (int j = 0; j < 4; ++j) o[j][r] *= corr[r];
  }
}

// ---------------------------------------------------------------------------
// All per-wave state for the attention main loop, processed per 32-key chunk.
// ---------------------------------------------------------------------------
struct AttnState {
  const unsigned short* Kh;
  const unsigned short* Vh;
  v16bf qa0, qa1, qb0, qb1;
  v8f oa[4], ob[4];
  float ma[8], la[8], mb[8], lb[8];
  unsigned short (*pLds)[16 * 32];
  int qbase, nn, hi;
};

template <bool DIAG>
DEV void attn_chunk(AttnState& st, int kc) {
  const v16bf k00 = load_frag(st.Kh, kc,      kD, 0);
  const v16bf k01 = load_frag(st.Kh, kc,      kD, 32);
  const v16bf k10 = load_frag(st.Kh, kc + 16, kD, 0);
  const v16bf k11 = load_frag(st.Kh, kc + 16, kD, 32);

  // S tiles: 32 rows x 32 keys in four 16x16 f32 accumulators
  v8f sa0 = {}, sa1 = {}, sb0 = {}, sb1 = {};
  sa0 = wmma_bf16(st.qa0, k00, sa0);  sa0 = wmma_bf16(st.qa1, k01, sa0);
  sa1 = wmma_bf16(st.qa0, k10, sa1);  sa1 = wmma_bf16(st.qa1, k11, sa1);
  sb0 = wmma_bf16(st.qb0, k00, sb0);  sb0 = wmma_bf16(st.qb1, k01, sb0);
  sb1 = wmma_bf16(st.qb0, k10, sb1);  sb1 = wmma_bf16(st.qb1, k11, sb1);

  softmax_update<DIAG>(sa0, sa1, st.ma, st.la, st.oa, st.qbase,      kc,
                       st.pLds[0], st.nn, st.hi);
  softmax_update<DIAG>(sb0, sb1, st.mb, st.lb, st.ob, st.qbase + 16, kc,
                       st.pLds[1], st.nn, st.hi);

  __syncthreads();  // DS store -> DS load ordering (single-wave WG)

  // re-read both P tiles as 16x32 A-fragments (ds_load_b128 x2 per tile)
  union { v16bf v; u32x4 q[2]; } pfa, pfb;
  {
    const unsigned short* pp = &st.pLds[0][st.nn * 32 + st.hi * 8];
    pfa.q[0] = *(const u32x4*)(pp);
    pfa.q[1] = *(const u32x4*)(pp + 16);
    pp = &st.pLds[1][st.nn * 32 + st.hi * 8];
    pfb.q[0] = *(const u32x4*)(pp);
    pfb.q[1] = *(const u32x4*)(pp + 16);
  }
  __syncthreads();

  // O += P @ V : both row tiles share the same V fragments
#pragma unroll
  for (int j = 0; j < 4; ++j) {
    const v16bf vf = load_frag(st.Vh, j * 16, kT, kc);
    st.oa[j] = wmma_bf16(pfa.v, vf, st.oa[j]);
    st.ob[j] = wmma_bf16(pfb.v, vf, st.ob[j]);
  }
}

// ---------------------------------------------------------------------------
// Flash attention: one wave per (b,h, 32-row q-tile); the two 16-row subtiles
// share every K and V fragment. Causal: qbase is a multiple of 32, so only
// the peeled kc==qbase chunk carries masking code.
//   Q,K: [B*H, T, D] bf16   Vt: [B*H, D, T] bf16   Obf: [B, T, C] bf16
// ---------------------------------------------------------------------------
__global__ __launch_bounds__(32) void attn_kernel(
    const unsigned short* __restrict__ Q,
    const unsigned short* __restrict__ K,
    const unsigned short* __restrict__ Vt,
    unsigned short* __restrict__ Obf) {
  __shared__ __attribute__((aligned(16))) unsigned short pLds[2][16 * 32];

  const int bh    = blockIdx.y;
  const int qbase = blockIdx.x * 32;
  const int lane  = threadIdx.x & 31;

  AttnState st;
  st.Kh    = K  + (size_t)bh * kT * kD;
  st.Vh    = Vt + (size_t)bh * kD * kT;
  st.pLds  = pLds;
  st.qbase = qbase;
  st.nn    = lane & 15;
  st.hi    = lane >> 4;

  const unsigned short* Qh = Q + (size_t)bh * kT * kD;
  st.qa0 = load_frag(Qh, qbase,      kD, 0);
  st.qa1 = load_frag(Qh, qbase,      kD, 32);
  st.qb0 = load_frag(Qh, qbase + 16, kD, 0);
  st.qb1 = load_frag(Qh, qbase + 16, kD, 32);

#pragma unroll
  for (int j = 0; j < 4; ++j) { st.oa[j] = v8f{}; st.ob[j] = v8f{}; }
#pragma unroll
  for (int r = 0; r < 8; ++r) {
    st.ma[r] = -INFINITY; st.la[r] = 0.f;
    st.mb[r] = -INFINITY; st.lb[r] = 0.f;
  }

  // steady state: full (unmasked) chunks kc = 0 .. qbase-32
  for (int kc = 0; kc < qbase; kc += 32) {
    // prefetch next chunk's K rows and Vt columns (global_prefetch_b8)
    const int kn = kc + 32;
    __builtin_prefetch(st.Kh + (size_t)(kn + lane) * kD, 0, 3);
    __builtin_prefetch(st.Vh + (size_t)lane * kT + kn, 0, 3);
    __builtin_prefetch(st.Vh + (size_t)(lane + 32) * kT + kn, 0, 3);
    attn_chunk<false>(st, kc);
  }
  // peeled diagonal chunk (the only one with mask VALU)
  attn_chunk<true>(st, qbase);

  // normalize and store merged heads as bf16 [B, T, C]
  const int b = bh >> 4;
  const int h = bh & 15;
#pragma unroll
  for (int r = 0; r < 8; ++r) {
    const int ta = qbase + r + st.hi * 8;
    const int tb = ta + 16;
    const float inva = 1.f / st.la[r];
    const float invb = 1.f / st.lb[r];
    const size_t rowa = ((size_t)(b * kT + ta)) * kC + h * kD;
    const size_t rowb = ((size_t)(b * kT + tb)) * kC + h * kD;
#pragma unroll
    for (int j = 0; j < 4; ++j) {
      Obf[rowa + j * 16 + st.nn] = f32_to_bf16_bits(st.oa[j][r] * inva);
      Obf[rowb + j * 16 + st.nn] = f32_to_bf16_bits(st.ob[j][r] * invb);
    }
  }
}

// ---------------------------------------------------------------------------
// Launch
// ---------------------------------------------------------------------------
extern "C" void kernel_launch(void* const* d_in, const int* in_sizes, int n_in,
                              void* d_out, int out_size, void* d_ws, size_t ws_size,
                              hipStream_t stream) {
  (void)in_sizes; (void)n_in; (void)out_size; (void)ws_size;

  const float* x  = (const float*)d_in[0];
  const float* Wq = (const float*)d_in[1];
  const float* Wk = (const float*)d_in[2];
  const float* Wv = (const float*)d_in[3];
  const float* Wo = (const float*)d_in[4];

  char* ws = (char*)d_ws;
  const size_t xbytes = (size_t)kM * kC * 2;   // 16 MB activations bf16
  const size_t wbytes = (size_t)kC * kC * 2;   //  2 MB per weight bf16

  unsigned short* xbf = (unsigned short*)(ws);
  unsigned short* wq  = (unsigned short*)(ws + xbytes + 0 * wbytes);
  unsigned short* wk  = (unsigned short*)(ws + xbytes + 1 * wbytes);
  unsigned short* wv  = (unsigned short*)(ws + xbytes + 2 * wbytes);
  unsigned short* wo  = (unsigned short*)(ws + xbytes + 3 * wbytes);
  unsigned short* Qb  = (unsigned short*)(ws + xbytes + 4 * wbytes);
  unsigned short* Kb  = Qb + (size_t)kM * kC;
  unsigned short* Vb  = Kb + (size_t)kM * kC;
  unsigned short* Ab  = Vb + (size_t)kM * kC;  // attention output, bf16 [B,T,C]
  // total ws use: 16 + 8 + 64 MB = 88 MB

  cvt_bf16_kernel<<<2048, 256, 0, stream>>>(x,  xbf, kM * kC);
  cvt_bf16_kernel<<<512,  256, 0, stream>>>(Wq, wq,  kC * kC);
  cvt_bf16_kernel<<<512,  256, 0, stream>>>(Wk, wk,  kC * kC);
  cvt_bf16_kernel<<<512,  256, 0, stream>>>(Wv, wv,  kC * kC);
  cvt_bf16_kernel<<<512,  256, 0, stream>>>(Wo, wo,  kC * kC);

  const dim3 ggrid(kM / 128, kC / 64);         // 64 x 16 blocks, 128 thr
  // Q carries the 1/sqrt(D) softmax scale folded into its projection.
  gemm_kernel<1><<<ggrid, 128, 0, stream>>>(xbf, wq, Qb, 0.125f);
  gemm_kernel<1><<<ggrid, 128, 0, stream>>>(xbf, wk, Kb, 1.0f);
  gemm_kernel<2><<<ggrid, 128, 0, stream>>>(xbf, wv, Vb, 1.0f);

  attn_kernel<<<dim3(kT / 32, kB * kH), 32, 0, stream>>>(Qb, Kb, Vb, Ab);

  gemm_kernel<0><<<ggrid, 128, 0, stream>>>(Ab, wo, d_out, 1.0f);
}
